// QuantizeModel_47227460387394
// MI455X (gfx1250) — matrix operand order repaired and checked by reading, not compiled, not run on hardware
//
#include <hip/hip_runtime.h>
#include <hip/hip_bf16.h>

// Problem constants (from reference): B=8, T=2048 -> BT=16384 tokens
#define BT   16384
#define DDIM 256
#define FEAT 756
#define QN   8
#define KN   1024

typedef __attribute__((ext_vector_type(2)))  float  v2f;
typedef __attribute__((ext_vector_type(8)))  float  v8f;
typedef __attribute__((ext_vector_type(8)))  __bf16 v8bf;
typedef __attribute__((ext_vector_type(16))) __bf16 v16bf;

// ---------------------------------------------------------------------------
// Prep: bf16 copy of codebooks + bias = -0.5*||c||^2 per codeword.
// One wave per codebook row (256 f32). grid = 8192/8 blocks of 256 threads.
// ---------------------------------------------------------------------------
__global__ __launch_bounds__(256) void prep_cb(const float* __restrict__ cb,
                                               __bf16* __restrict__ cbBf,
                                               float* __restrict__ bias) {
  int wv = threadIdx.x >> 5, lane = threadIdx.x & 31;
  int row = blockIdx.x * 8 + wv;                 // 0 .. Q*K-1
  const float* src = cb + (size_t)row * DDIM;
  __bf16* dst = cbBf + (size_t)row * DDIM;
  float ss = 0.f;
  for (int i = 0; i < 8; ++i) {
    int e = lane + 32 * i;
    float v = src[e];
    ss += v * v;
    dst[e] = (__bf16)v;
  }
  for (int off = 16; off >= 1; off >>= 1) ss += __shfl_xor(ss, off, 32);
  if (lane == 0) bias[row] = -0.5f * ss;
}

__global__ void zero_qsum(float4* __restrict__ qsum) {
  qsum[(size_t)blockIdx.x * blockDim.x + threadIdx.x] = make_float4(0.f, 0.f, 0.f, 0.f);
}

// ---------------------------------------------------------------------------
// GEMM 1: residual = x @ w_in, with the (B,T,252,3)->(B,T,756) transpose
// fused into the A-operand gather. f32 WMMA 16x16x4, K=756 (189 steps).
// One wave per 16x16 output tile; 8 waves / block.
// ---------------------------------------------------------------------------
__global__ __launch_bounds__(256) void gemm_in(const float* __restrict__ x,
                                               const float* __restrict__ w_in,
                                               float* __restrict__ residual) {
  int wv = threadIdx.x >> 5, lane = threadIdx.x & 31;
  int g = lane >> 4, nlow = lane & 15;
  int wt = blockIdx.x * 8 + wv;                  // 0 .. 16383
  int mt = wt >> 4, nt = wt & 15;
  int tokA = mt * 16 + nlow;                     // A-row token (M = lane%16)
  int ncol = nt * 16 + nlow;
  v8f acc = {};
  for (int k0 = 0; k0 < FEAT; k0 += 4) {
    int f0 = k0 + 2 * g;                         // A VGPR0/1 hold K = 2g, 2g+1
    int f1 = f0 + 1;
    // feature f = c*252 + r  maps to input element [tok][r][c]
    int c0 = (f0 >= 504) ? 2 : ((f0 >= 252) ? 1 : 0);
    int c1 = (f1 >= 504) ? 2 : ((f1 >= 252) ? 1 : 0);
    v2f a, b;
    a.x = x[((size_t)tokA * 252 + (f0 - c0 * 252)) * 3 + c0];
    a.y = x[((size_t)tokA * 252 + (f1 - c1 * 252)) * 3 + c1];
    b.x = w_in[(size_t)f0 * DDIM + ncol];
    b.y = w_in[(size_t)f1 * DDIM + ncol];
    acc = __builtin_amdgcn_wmma_f32_16x16x4_f32(false, a, false, b,
                                                (short)0, acc, false, false);
  }
  int base = mt * 16;
  for (int v = 0; v < 8; ++v)
    residual[(size_t)(base + v + 8 * g) * DDIM + ncol] = acc[v];
}

// ---------------------------------------------------------------------------
// One residual-VQ step (launched 8x, sequential on stream).
// Block = 128 threads (4 waves), 16 tokens per block.
// Score S = residual_bf16 @ cb_bf16^T + bias  via v_wmma_f32_16x16x32_bf16,
// row-argmax (shfl_xor + LDS), then exact-f32 gather/update.
// ---------------------------------------------------------------------------
__global__ __launch_bounds__(128) void vq_step(int q,
    const float*  __restrict__ cb,       // original f32 codebooks [Q,K,D]
    const __bf16* __restrict__ cbBf,     // bf16 codebooks
    const float*  __restrict__ bias,     // -0.5*||c||^2  [Q,K]
    float* __restrict__ residual,
    float* __restrict__ qsum,
    float* __restrict__ all_q,           // [Q,BT,D] section of d_out
    float* __restrict__ encIdx) {        // [BT,Q] section of d_out
  __shared__ __bf16 aTile[16 * 256];     // 16 tokens x 256 dims, bf16 (8 KB)
  __shared__ float  redVal[16][4];
  __shared__ int    redIdx[16][4];
  __shared__ int    idxSh[16];

  int tid  = threadIdx.x;
  int tok0 = blockIdx.x * 16;

  // Stage residual tile as bf16 in LDS (coalesced f32 reads).
  for (int e = tid; e < 4096; e += 128) {
    int m = e >> 8, d = e & 255;
    aTile[e] = (__bf16)residual[(size_t)(tok0 + m) * DDIM + d];
  }
  __syncthreads();

  int wv = tid >> 5, lane = tid & 31, g = lane >> 4, nlow = lane & 15;

  // A fragments (16x32 bf16 per K-chunk), per ISA layout:
  // halves 0..7 -> K = kc*32 + 8g + (0..7); halves 8..15 -> +16.
  v16bf aF[8];
  {
    int m = lane & 15;
    for (int kc = 0; kc < 8; ++kc) {
      const __bf16* p = &aTile[m * 256 + kc * 32 + 8 * g];
      v8bf lo = *(const v8bf*)p;
      v8bf hi = *(const v8bf*)(p + 16);
      aF[kc] = __builtin_shufflevector(lo, hi, 0, 1, 2, 3, 4, 5, 6, 7,
                                       8, 9, 10, 11, 12, 13, 14, 15);
    }
  }

  float bv[8];
  int   bi[8];
  for (int v = 0; v < 8; ++v) { bv[v] = -3.4e38f; bi[v] = 0; }

  const __bf16* cbq   = cbBf + (size_t)q * KN * DDIM;
  const float*  biasq = bias + q * KN;

  // Each wave covers 16 of the 64 N-tiles (1024 codewords total).
  for (int j = 0; j < 16; ++j) {
    int nt = wv + 4 * j;
    int n  = nt * 16 + nlow;                     // this lane's codeword column
    float bsc = biasq[n];
    v8f acc = { bsc, bsc, bsc, bsc, bsc, bsc, bsc, bsc };
    // B 32x16 bf16: lane holds halves K = kc*32 + 8g + (0..7) and +16 of codeword row n.
    const __bf16* cbrow = cbq + (size_t)n * DDIM + 8 * g;
    for (int kc = 0; kc < 8; ++kc) {
      v8bf blo = *(const v8bf*)(cbrow + kc * 32);
      v8bf bhi = *(const v8bf*)(cbrow + kc * 32 + 16);
      v16bf b = __builtin_shufflevector(blo, bhi, 0, 1, 2, 3, 4, 5, 6, 7,
                                        8, 9, 10, 11, 12, 13, 14, 15);
      acc = __builtin_amdgcn_wmma_f32_16x16x32_bf16(false, aF[kc], false, b,
                                                    (short)0, acc, false, false);
    }
    for (int v = 0; v < 8; ++v)
      if (acc[v] > bv[v]) { bv[v] = acc[v]; bi[v] = n; }
  }

  // Cross-lane argmax within each 16-lane half (same M rows, different N).
  for (int v = 0; v < 8; ++v) {
    float val = bv[v];
    int   id  = bi[v];
    for (int off = 1; off <= 8; off <<= 1) {
      float ov = __shfl_xor(val, off, 32);
      int   oi = __shfl_xor(id, off, 32);
      if (ov > val || (ov == val && oi < id)) { val = ov; id = oi; }
    }
    if (nlow == 0) { int m = v + 8 * g; redVal[m][wv] = val; redIdx[m][wv] = id; }
  }
  __syncthreads();

  // Combine the 4 waves' candidates; emit encoding index.
  if (tid < 16) {
    float best = redVal[tid][0];
    int   bid  = redIdx[tid][0];
    for (int w = 1; w < 4; ++w) {
      float v = redVal[tid][w];
      int   i = redIdx[tid][w];
      if (v > best || (v == best && i < bid)) { best = v; bid = i; }
    }
    idxSh[tid] = bid;
    encIdx[(size_t)(tok0 + tid) * QN + q] = (float)bid;
  }
  __syncthreads();

  // Exact f32 gather + residual/qsum update + all_q output.
  for (int e = tid; e < 4096; e += 128) {
    int m = e >> 8, d = e & 255;
    size_t off = (size_t)(tok0 + m) * DDIM + d;
    float c = cb[((size_t)q * KN + idxSh[m]) * DDIM + d];
    float r = residual[off] - c;
    residual[off] = r;
    qsum[off] += c;
    all_q[(size_t)q * BT * DDIM + off] = c;
  }
}

// ---------------------------------------------------------------------------
// GEMM 3: quantized = relu(qsum @ w_out). f32 WMMA 16x16x4, K=256 (64 steps).
// N=756 -> 48 N-tiles, last tile store-masked.
// ---------------------------------------------------------------------------
__global__ __launch_bounds__(256) void gemm_out(const float* __restrict__ qsum,
                                                const float* __restrict__ w_out,
                                                float* __restrict__ out) {
  int wv = threadIdx.x >> 5, lane = threadIdx.x & 31;
  int g = lane >> 4, nlow = lane & 15;
  int wt = blockIdx.x * 8 + wv;                  // 0 .. 49151
  int mt = wt / 48, nt = wt - mt * 48;
  int tokA = mt * 16 + nlow;
  int ncol = nt * 16 + nlow;
  v8f acc = {};
  for (int k0 = 0; k0 < DDIM; k0 += 4) {
    int d0 = k0 + 2 * g;
    v2f a = *(const v2f*)(qsum + (size_t)tokA * DDIM + d0);
    v2f b;
    b.x = w_out[(size_t)d0 * FEAT + ncol];
    b.y = w_out[(size_t)(d0 + 1) * FEAT + ncol];
    acc = __builtin_amdgcn_wmma_f32_16x16x4_f32(false, a, false, b,
                                                (short)0, acc, false, false);
  }
  if (ncol < FEAT) {                             // ragged edge of N=756
    int base = mt * 16;
    for (int v = 0; v < 8; ++v) {
      float r = acc[v];
      out[(size_t)(base + v + 8 * g) * FEAT + ncol] = r > 0.f ? r : 0.f;
    }
  }
}

// ---------------------------------------------------------------------------
extern "C" void kernel_launch(void* const* d_in, const int* in_sizes, int n_in,
                              void* d_out, int out_size, void* d_ws, size_t ws_size,
                              hipStream_t stream) {
  const float* inputs    = (const float*)d_in[0];   // [8,2048,252,3]
  const float* codebooks = (const float*)d_in[1];   // [8,1024,256]
  const float* w_in      = (const float*)d_in[2];   // [756,256]
  const float* w_out     = (const float*)d_in[3];   // [256,756]

  float* out       = (float*)d_out;
  float* quantized = out;                                   // BT*756
  float* all_q     = out + (size_t)BT * FEAT;               // Q*BT*256
  float* encIdx    = all_q + (size_t)QN * BT * DDIM;        // BT*Q

  // Workspace: residual (16 MB) | qsum (16 MB) | cb bf16 (4 MB) | bias (32 KB)
  float*  residual = (float*)d_ws;
  float*  qsum     = residual + (size_t)BT * DDIM;
  __bf16* cbBf     = (__bf16*)(qsum + (size_t)BT * DDIM);
  float*  bias     = (float*)(cbBf + (size_t)QN * KN * DDIM);

  prep_cb<<<(QN * KN) / 8, 256, 0, stream>>>(codebooks, cbBf, bias);
  zero_qsum<<<(BT * DDIM) / (256 * 4), 256, 0, stream>>>((float4*)qsum);
  gemm_in<<<(BT / 16) * (DDIM / 16) / 8, 256, 0, stream>>>(inputs, w_in, residual);
  for (int q = 0; q < QN; ++q)
    vq_step<<<BT / 16, 128, 0, stream>>>(q, codebooks, cbBf, bias,
                                         residual, qsum, all_q, encIdx);
  gemm_out<<<(BT / 16) * 48 / 8, 256, 0, stream>>>(qsum, w_out, quantized);
}
